// MSA_4638564680551
// MI455X (gfx1250) — matrix operand order repaired and checked by reading, not compiled, not run on hardware
//
#include <hip/hip_runtime.h>
#include <math.h>

typedef __attribute__((ext_vector_type(16))) __bf16 v16bf;
typedef __attribute__((ext_vector_type(8)))  __bf16 v8bf;
typedef __attribute__((ext_vector_type(4)))  __bf16 v4bf;
typedef __attribute__((ext_vector_type(8)))  float  v8f;
typedef __attribute__((ext_vector_type(4)))  float  f32x4;
typedef __attribute__((ext_vector_type(4)))  unsigned int u32x4;
typedef __attribute__((ext_vector_type(2)))  unsigned int u32x2;

union V16U { v16bf v; u32x4 q[2]; };
union V8U  { v8bf  v; u32x4 q;    };
union V4U  { v4bf  v; u32x2 q;    };

__device__ __forceinline__ v8f vzero8() {
    v8f z = {0.f,0.f,0.f,0.f,0.f,0.f,0.f,0.f};
    return z;
}

// A fragment: 16x32 bf16, row m = lane&15, K halves per ISA table
__device__ __forceinline__ v16bf load_a_frag(const __bf16* A, int lda, int row0, int k0, int lane) {
    int m = lane & 15, half = lane >> 4;
    const __bf16* p = A + (size_t)(row0 + m) * lda + k0 + half * 8;
    V16U r;
    r.q[0] = *(const u32x4*)p;        // K = k0 + half*8 + [0..7]
    r.q[1] = *(const u32x4*)(p + 16); // K = k0 + 16 + half*8 + [0..7]
    return r.v;
}

// B fragment: 32x16 bf16 from K-major (transposed) storage Bt[n][k]
__device__ __forceinline__ v16bf load_b_frag(const __bf16* Bt, int ldb, int col0, int k0, int lane) {
    int n = lane & 15, half = lane >> 4;
    const __bf16* p = Bt + (size_t)(col0 + n) * ldb + k0 + half * 8;
    V16U r;
    r.q[0] = *(const u32x4*)p;        // K = k0 + half*8 + [0..7]
    r.q[1] = *(const u32x4*)(p + 16); // K = k0 + 16 + half*8 + [0..7]
    return r.v;
}

#define WMMA_BF16(a, b, c) \
    __builtin_amdgcn_wmma_f32_16x16x32_bf16(false, (a), false, (b), (short)0, (c), false, false)

__device__ __forceinline__ void store8_bf16(__bf16* p, v8f acc) {
    V8U pk;
#pragma unroll
    for (int r = 0; r < 8; ++r) pk.v[r] = (__bf16)acc[r];
    *(u32x4*)p = pk.q;
}

// ---------------- elementwise conversions ----------------

__global__ __launch_bounds__(256) void cvt_x_kernel(const float* __restrict__ x,
                                                    __bf16* __restrict__ xb) {
    size_t i = ((size_t)blockIdx.x * 256 + threadIdx.x) * 4;
    f32x4 v = *(const f32x4*)(x + i);
    V4U pk;
#pragma unroll
    for (int r = 0; r < 4; ++r) pk.v[r] = (__bf16)v[r];
    *(u32x2*)(xb + i) = pk.q;
}

// Build WqkvT[1536][512] and WpT[512][512] (bf16, K-major) from row-major f32 weights.
__global__ __launch_bounds__(256) void wtrans_kernel(const float* __restrict__ Wq,
                                                     const float* __restrict__ Wk,
                                                     const float* __restrict__ Wv,
                                                     const float* __restrict__ Wp,
                                                     __bf16* __restrict__ WqkvT,
                                                     __bf16* __restrict__ WpT) {
    int row = blockIdx.x; // 0..2047
    if (row < 1536) {
        const float* W = (row < 512) ? Wq : (row < 1024) ? Wk : Wv;
        int j = row & 511;
        __bf16* dst = WqkvT + (size_t)row * 512;
        for (int k = threadIdx.x; k < 512; k += 256) dst[k] = (__bf16)W[(size_t)k * 512 + j];
    } else {
        int j = row - 1536;
        __bf16* dst = WpT + (size_t)j * 512;
        for (int k = threadIdx.x; k < 512; k += 256) dst[k] = (__bf16)Wp[(size_t)k * 512 + j];
    }
}

// ---------------- QKV projection GEMM (M=65536, N=1536, K=512) ----------------
// Q,K stored transposed [b*8*64][16384]; V stored [token][512] bf16.
__global__ __launch_bounds__(256) void qkv_gemm_kernel(const __bf16* __restrict__ Xbf,
                                                       const __bf16* __restrict__ WqkvT,
                                                       __bf16* __restrict__ Qt,
                                                       __bf16* __restrict__ Kt,
                                                       __bf16* __restrict__ Vbf) {
    int lane = threadIdx.x & 31;
    int wave = threadIdx.x >> 5;
    int waveM = wave & 3, waveN = wave >> 2;       // 4 x 2 wave grid
    int row0 = blockIdx.x * 128 + waveM * 32;      // tokens
    int col0 = blockIdx.y * 128 + waveN * 64;      // output channels (qkv concat)

    v8f acc[2][4];
#pragma unroll
    for (int i = 0; i < 2; ++i)
#pragma unroll
        for (int j = 0; j < 4; ++j) acc[i][j] = vzero8();

    for (int k0 = 0; k0 < 512; k0 += 32) {
        v16bf a[2], bf[4];
#pragma unroll
        for (int i = 0; i < 2; ++i) a[i] = load_a_frag(Xbf, 512, row0 + i * 16, k0, lane);
#pragma unroll
        for (int j = 0; j < 4; ++j) bf[j] = load_b_frag(WqkvT, 512, col0 + j * 16, k0, lane);
#pragma unroll
        for (int i = 0; i < 2; ++i)
#pragma unroll
            for (int j = 0; j < 4; ++j) acc[i][j] = WMMA_BF16(a[i], bf[j], acc[i][j]);
    }

    int nn = lane & 15, half = lane >> 4;
#pragma unroll
    for (int i = 0; i < 2; ++i)
#pragma unroll
        for (int j = 0; j < 4; ++j) {
            int col = col0 + j * 16 + nn;
            int gr0 = row0 + i * 16 + half * 8; // 8 consecutive tokens along vgpr index
            if (col < 1024) {
                __bf16* base = (col < 512) ? Qt : Kt;
                int ch = col & 511;          // h*64 + d
                int bb = gr0 >> 14;
                int tok = gr0 & 16383;
                store8_bf16(base + ((size_t)(bb * 512 + ch)) * 16384 + tok, acc[i][j]);
            } else {
                int cch = col - 1024;
#pragma unroll
                for (int r = 0; r < 8; ++r)
                    Vbf[(size_t)(gr0 + r) * 512 + cch] = (__bf16)acc[i][j][r];
            }
        }
}

// ---------------- token-dim L2 norms of Q,K ----------------
__global__ __launch_bounds__(256) void norms_kernel(const __bf16* __restrict__ Qt,
                                                    const __bf16* __restrict__ Kt,
                                                    float* __restrict__ nq,
                                                    float* __restrict__ nk) {
    const __bf16* src = blockIdx.y ? Kt : Qt;
    float* dst = blockIdx.y ? nk : nq;
    const __bf16* p = src + (size_t)blockIdx.x * 16384;
    float s = 0.f;
    for (int i = threadIdx.x; i < 16384; i += 256) {
        float v = (float)p[i];
        s += v * v;
    }
    __shared__ float red[256];
    red[threadIdx.x] = s;
    __syncthreads();
    for (int off = 128; off > 0; off >>= 1) {
        if (threadIdx.x < off) red[threadIdx.x] += red[threadIdx.x + off];
        __syncthreads();
    }
    if (threadIdx.x == 0) dst[blockIdx.x] = sqrtf(red[0]);
}

// ---------------- raw channel-attention dots: S[bh][d][e] = sum_n K[d,n]*Q[e,n] ----------------
__global__ __launch_bounds__(256) void attn_dots_kernel(const __bf16* __restrict__ Qt,
                                                        const __bf16* __restrict__ Kt,
                                                        float* __restrict__ S) {
    __shared__ float sS[64 * 64];
    for (int i = threadIdx.x; i < 4096; i += 256) sS[i] = 0.f;
    __syncthreads();

    int bh = blockIdx.x;
    const __bf16* Kb = Kt + (size_t)bh * 64 * 16384;
    const __bf16* Qb = Qt + (size_t)bh * 64 * 16384;
    int lane = threadIdx.x & 31;
    int wave = threadIdx.x >> 5;
    int ih = wave >> 2;       // M half (32 rows of d)
    int ns = wave & 3;        // token slice
    int nbase = blockIdx.y * 2048 + ns * 512;

    v8f acc[2][4];
#pragma unroll
    for (int i = 0; i < 2; ++i)
#pragma unroll
        for (int j = 0; j < 4; ++j) acc[i][j] = vzero8();

    for (int k0 = nbase; k0 < nbase + 512; k0 += 32) {
        v16bf a[2], bf[4];
#pragma unroll
        for (int i = 0; i < 2; ++i) a[i] = load_a_frag(Kb, 16384, (ih * 2 + i) * 16, k0, lane);
#pragma unroll
        for (int j = 0; j < 4; ++j) bf[j] = load_b_frag(Qb, 16384, j * 16, k0, lane);
#pragma unroll
        for (int i = 0; i < 2; ++i)
#pragma unroll
            for (int j = 0; j < 4; ++j) acc[i][j] = WMMA_BF16(a[i], bf[j], acc[i][j]);
    }

    int nn = lane & 15, half = lane >> 4;
#pragma unroll
    for (int i = 0; i < 2; ++i)
#pragma unroll
        for (int j = 0; j < 4; ++j)
#pragma unroll
            for (int r = 0; r < 8; ++r) {
                int d = (ih * 2 + i) * 16 + half * 8 + r;
                int e = j * 16 + nn;
                atomicAdd(&sS[d * 64 + e], acc[i][j][r]);
            }
    __syncthreads();
    float* Sg = S + (size_t)bh * 4096;
    for (int i = threadIdx.x; i < 4096; i += 256) atomicAdd(&Sg[i], sS[i]);
}

// ---------------- softmax with cosine normalization; emits attnT[bh][e][d] bf16 ----------------
__global__ __launch_bounds__(64) void softmax_kernel(const float* __restrict__ S,
                                                     const float* __restrict__ nq,
                                                     const float* __restrict__ nk,
                                                     const float* __restrict__ rescale,
                                                     __bf16* __restrict__ attnT) {
    int bh = blockIdx.x;
    int h = bh & 7;
    int d = threadIdx.x;
    const float* Sr = S + (size_t)bh * 4096 + d * 64;
    float rs = rescale[h];
    float invk = rs / nk[bh * 64 + d];
    float vals[64];
    float mx = -3.4e38f;
#pragma unroll
    for (int e = 0; e < 64; ++e) {
        float v = Sr[e] * invk / nq[bh * 64 + e];
        vals[e] = v;
        mx = fmaxf(mx, v);
    }
    float sum = 0.f;
#pragma unroll
    for (int e = 0; e < 64; ++e) {
        float v = __expf(vals[e] - mx);
        vals[e] = v;
        sum += v;
    }
    float inv = 1.f / sum;
    __bf16* dst = attnT + (size_t)bh * 4096 + d;
#pragma unroll
    for (int e = 0; e < 64; ++e) dst[e * 64] = (__bf16)(vals[e] * inv);
}

// ---------------- Wp_eff[b][e_glob][c'] = sum_d attn[h][d][e] * Wp[h*64+d][c'], stored T ----------------
__global__ __launch_bounds__(256) void wpeff_kernel(const __bf16* __restrict__ attnT,
                                                    const __bf16* __restrict__ WpT,
                                                    __bf16* __restrict__ WpeT) {
    int bh = blockIdx.x >> 1;
    int nhalf = blockIdx.x & 1;
    int b = bh >> 3, h = bh & 7;
    int lane = threadIdx.x & 31;
    int wave = threadIdx.x >> 5;
    const __bf16* At = attnT + (size_t)bh * 4096; // [e][d], lda = 64
    int col0 = nhalf * 256 + wave * 32;

    v8f acc[4][2];
#pragma unroll
    for (int i = 0; i < 4; ++i)
#pragma unroll
        for (int j = 0; j < 2; ++j) acc[i][j] = vzero8();

    for (int k0 = 0; k0 < 64; k0 += 32) {
        v16bf a[4], bf[2];
#pragma unroll
        for (int i = 0; i < 4; ++i) a[i] = load_a_frag(At, 64, i * 16, k0, lane);
#pragma unroll
        for (int j = 0; j < 2; ++j)
            bf[j] = load_b_frag(WpT + h * 64, 512, col0 + j * 16, k0, lane);
#pragma unroll
        for (int i = 0; i < 4; ++i)
#pragma unroll
            for (int j = 0; j < 2; ++j) acc[i][j] = WMMA_BF16(a[i], bf[j], acc[i][j]);
    }

    int nn = lane & 15, half = lane >> 4;
    __bf16* dst = WpeT + (size_t)b * 512 * 512;
#pragma unroll
    for (int i = 0; i < 4; ++i)
#pragma unroll
        for (int j = 0; j < 2; ++j) {
            int cp = col0 + j * 16 + nn;
            int e0 = i * 16 + half * 8;
            store8_bf16(dst + (size_t)cp * 512 + h * 64 + e0, acc[i][j]);
        }
}

// ---------------- out_c = V @ Wp_eff[b] + bp  (M=65536, N=512, K=512) ----------------
__global__ __launch_bounds__(256) void outc_gemm_kernel(const __bf16* __restrict__ Vbf,
                                                        const __bf16* __restrict__ WpeT,
                                                        const float* __restrict__ bp,
                                                        float* __restrict__ out) {
    int lane = threadIdx.x & 31;
    int wave = threadIdx.x >> 5;
    int waveM = wave & 3, waveN = wave >> 2;
    int row0 = blockIdx.x * 128 + waveM * 32;
    int col0 = blockIdx.y * 128 + waveN * 64;
    int bb = (blockIdx.x * 128) >> 14;
    const __bf16* Bt = WpeT + (size_t)bb * 512 * 512;

    v8f acc[2][4];
#pragma unroll
    for (int i = 0; i < 2; ++i)
#pragma unroll
        for (int j = 0; j < 4; ++j) acc[i][j] = vzero8();

    for (int k0 = 0; k0 < 512; k0 += 32) {
        v16bf a[2], bf[4];
#pragma unroll
        for (int i = 0; i < 2; ++i) a[i] = load_a_frag(Vbf, 512, row0 + i * 16, k0, lane);
#pragma unroll
        for (int j = 0; j < 4; ++j) bf[j] = load_b_frag(Bt, 512, col0 + j * 16, k0, lane);
#pragma unroll
        for (int i = 0; i < 2; ++i)
#pragma unroll
            for (int j = 0; j < 4; ++j) acc[i][j] = WMMA_BF16(a[i], bf[j], acc[i][j]);
    }

    int nn = lane & 15, half = lane >> 4;
#pragma unroll
    for (int i = 0; i < 2; ++i)
#pragma unroll
        for (int j = 0; j < 4; ++j) {
            int col = col0 + j * 16 + nn;
            int gr0 = row0 + i * 16 + half * 8;
            float bias = bp[col];
#pragma unroll
            for (int r = 0; r < 8; ++r)
                out[(size_t)(gr0 + r) * 512 + col] = acc[i][j][r] + bias;
        }
}

// ---------------- fused depthwise conv3x3 -> gelu(erf) -> conv3x3, accumulate into out -------
__global__ __launch_bounds__(256) void dwconv_kernel(const __bf16* __restrict__ Vbf,
                                                     const float* __restrict__ w1g,
                                                     const float* __restrict__ w2g,
                                                     float* __restrict__ out) {
    __shared__ float s1[12 * 12 * 64];
    __shared__ float s2[10 * 10 * 64];
    int b = blockIdx.z;
    int cs = blockIdx.y * 64;
    int ty = (blockIdx.x >> 4) * 8, tx = (blockIdx.x & 15) * 8;
    int cc = threadIdx.x & 63;
    int c = cs + cc;
    int pw = threadIdx.x >> 6; // 0..3 pixel workers

    // load 12x12 halo patch (zero padded at image edges)
    for (int p = pw; p < 144; p += 4) {
        int py = p / 12, px = p % 12;
        int gy = ty + py - 2, gx = tx + px - 2;
        float v = 0.f;
        if (gy >= 0 && gy < 128 && gx >= 0 && gx < 128)
            v = (float)Vbf[((size_t)(b * 16384 + gy * 128 + gx)) * 512 + c];
        s1[p * 64 + cc] = v;
    }
    __syncthreads();

    float w[9];
#pragma unroll
    for (int t = 0; t < 9; ++t) w[t] = w1g[c * 9 + t];

    // conv1 + exact gelu on the 10x10 intermediate; zero where the intermediate
    // lies outside the image (it is zero-padded input to conv2 in the reference)
    for (int p = pw; p < 100; p += 4) {
        int py = p / 10, px = p % 10;
        int gy1 = ty + py - 1, gx1 = tx + px - 1;
        float s = 0.f;
        if (gy1 >= 0 && gy1 < 128 && gx1 >= 0 && gx1 < 128) {
#pragma unroll
            for (int ky = 0; ky < 3; ++ky)
#pragma unroll
                for (int kx = 0; kx < 3; ++kx)
                    s += s1[((py + ky) * 12 + (px + kx)) * 64 + cc] * w[ky * 3 + kx];
            s = 0.5f * s * (1.f + erff(s * 0.70710678118654752f));
        }
        s2[p * 64 + cc] = s;
    }
    __syncthreads();

#pragma unroll
    for (int t = 0; t < 9; ++t) w[t] = w2g[c * 9 + t];

    for (int p = pw; p < 64; p += 4) {
        int py = p / 8, px = p % 8;
        float s = 0.f;
#pragma unroll
        for (int ky = 0; ky < 3; ++ky)
#pragma unroll
            for (int kx = 0; kx < 3; ++kx)
                s += s2[((py + ky) * 10 + (px + kx)) * 64 + cc] * w[ky * 3 + kx];
        int gy = ty + py, gx = tx + px;
        size_t o = ((size_t)(b * 16384 + gy * 128 + gx)) * 512 + c;
        out[o] += s;
    }
}

// ---------------- host-side launch ----------------

extern "C" void kernel_launch(void* const* d_in, const int* in_sizes, int n_in,
                              void* d_out, int out_size, void* d_ws, size_t ws_size,
                              hipStream_t stream) {
    const float* x_in    = (const float*)d_in[0];
    const float* Wq      = (const float*)d_in[1];
    const float* Wk      = (const float*)d_in[2];
    const float* Wv      = (const float*)d_in[3];
    const float* rescale = (const float*)d_in[4];
    const float* Wp      = (const float*)d_in[5];
    const float* bp      = (const float*)d_in[6];
    const float* c1w     = (const float*)d_in[7];
    const float* c2w     = (const float*)d_in[8];
    float* out = (float*)d_out;

    char* ws = (char*)d_ws;
    const size_t SZ_XBF   = 65536ull * 512 * 2;
    const size_t SZ_QT    = 4ull * 8 * 64 * 16384 * 2;
    const size_t SZ_KT    = SZ_QT;
    const size_t SZ_VBF   = 65536ull * 512 * 2;
    const size_t SZ_WQKVT = 1536ull * 512 * 2;
    const size_t SZ_WPT   = 512ull * 512 * 2;
    const size_t SZ_S     = 32ull * 64 * 64 * 4;
    const size_t SZ_NQ    = 32ull * 64 * 4;
    const size_t SZ_NK    = SZ_NQ;
    const size_t SZ_ATTNT = 32ull * 64 * 64 * 2;
    const size_t SZ_WPET  = 4ull * 512 * 512 * 2;

    size_t off = 0;
    __bf16* Xbf   = (__bf16*)(ws + off); off += SZ_XBF;
    __bf16* Qt    = (__bf16*)(ws + off); off += SZ_QT;
    __bf16* Kt    = (__bf16*)(ws + off); off += SZ_KT;
    __bf16* Vbf   = (__bf16*)(ws + off); off += SZ_VBF;
    __bf16* WqkvT = (__bf16*)(ws + off); off += SZ_WQKVT;
    __bf16* WpT   = (__bf16*)(ws + off); off += SZ_WPT;
    float*  Sbuf  = (float*) (ws + off); off += SZ_S;
    float*  nq    = (float*) (ws + off); off += SZ_NQ;
    float*  nk    = (float*) (ws + off); off += SZ_NK;
    __bf16* attnT = (__bf16*)(ws + off); off += SZ_ATTNT;
    __bf16* WpeT  = (__bf16*)(ws + off); off += SZ_WPET;
    if (ws_size < off) return;

    // 1) convert inputs / weights to bf16 (weights stored K-major for B fragments)
    cvt_x_kernel<<<dim3(32768), dim3(256), 0, stream>>>(x_in, Xbf);
    wtrans_kernel<<<dim3(2048), dim3(256), 0, stream>>>(Wq, Wk, Wv, Wp, WqkvT, WpT);

    // 2) QKV projection GEMM (WMMA bf16)
    qkv_gemm_kernel<<<dim3(512, 12), dim3(256), 0, stream>>>(Xbf, WqkvT, Qt, Kt, Vbf);

    // 3) attention dots (needs zeroed accumulator) + norms
    hipMemsetAsync(Sbuf, 0, SZ_S, stream);
    norms_kernel<<<dim3(2048, 2), dim3(256), 0, stream>>>(Qt, Kt, nq, nk);
    attn_dots_kernel<<<dim3(32, 8), dim3(256), 0, stream>>>(Qt, Kt, Sbuf);

    // 4) normalize + softmax -> attn^T (bf16)
    softmax_kernel<<<dim3(32), dim3(64), 0, stream>>>(Sbuf, nq, nk, rescale, attnT);

    // 5) fold attention into projection: Wp_eff[b] = attn^T @ Wp (WMMA)
    wpeff_kernel<<<dim3(64), dim3(256), 0, stream>>>(attnT, WpT, WpeT);

    // 6) out_c = V @ Wp_eff[b] + bp (WMMA)
    outc_gemm_kernel<<<dim3(512, 4), dim3(256), 0, stream>>>(Vbf, WpeT, bp, out);

    // 7) positional branch: fused dwconv -> gelu -> dwconv, accumulate into out
    dwconv_kernel<<<dim3(256, 8, 4), dim3(256), 0, stream>>>(Vbf, c1w, c2w, out);
}
